// MPNNLayer_10771777978619
// MI455X (gfx1250) — matrix operand order repaired and checked by reading, not compiled, not run on hardware
//
#include <hip/hip_runtime.h>
#include <hip/hip_bf16.h>

typedef __attribute__((ext_vector_type(16))) _Float16 v16h;
typedef __attribute__((ext_vector_type(8)))  _Float16 v8h;
typedef __attribute__((ext_vector_type(4)))  _Float16 v4h;
typedef __attribute__((ext_vector_type(8)))  float    v8f;

#define ND 64
#define ED 1
#define HD 128
#define NNODES 100000
#define NEDGES 1600000
#define TILE_E 128
#define TILE_N 128
#define LSTRIDE 136   // 128 + 8 halves pad -> conflict-free LDS rows
#define GSTRIDE 72    // 64 + 8 halves pad

__device__ __forceinline__ v8f wmma16(v16h a, v16h b, v8f c) {
  return __builtin_amdgcn_wmma_f32_16x16x32_f16(false, a, false, b,
                                                (short)0, c, false, false);
}

// CDNA5 async global->LDS copy (16B per lane), tracked by ASYNCcnt.
// vdst = LDS byte offset (low 32 bits of the generic shared pointer),
// vaddr = 64-bit global address, GV mode (no saddr).
__device__ __forceinline__ void async_ld16(void* ldsDst, const void* gSrc) {
  unsigned int l = (unsigned int)(unsigned long long)ldsDst;
  asm volatile("global_load_async_to_lds_b128 %0, %1, off"
               :: "v"(l), "v"(gSrc) : "memory");
}
__device__ __forceinline__ void wait_async() {
  asm volatile("s_wait_asynccnt 0x0" ::: "memory");
}

__device__ __forceinline__ float fast_sigmoid(float t) {
  return 1.f / (1.f + __expf(-t));
}
__device__ __forceinline__ float fast_tanh(float t) {
  return 1.f - 2.f / (1.f + __expf(2.f * t));
}

// One-time f32 -> f16 copy of the node table (each node reused ~16x by edges).
__global__ void cvt_x_f16(_Float16* __restrict__ xh, const float* __restrict__ x) {
  long i = (long)blockIdx.x * blockDim.x + threadIdx.x;
  if (i < (long)NNODES * ND / 4) {
    float4 v = ((const float4*)x)[i];
    v4h h; h[0] = (_Float16)v.x; h[1] = (_Float16)v.y;
    h[2] = (_Float16)v.z; h[3] = (_Float16)v.w;
    ((v4h*)xh)[i] = h;
  }
}

// Pack W [Nout x Kin, row-major] into per-lane WMMA B fragments (f16).
// B[k][n] = W[n][k].  ISA B layout: lane l -> N = l&15 within the 16-N tile;
// half j -> K = 8*(l>>4) + (j&7) + 16*(j>>3) within a 32-K tile.
__global__ void pack_weights(_Float16* __restrict__ dst,
                             const float* __restrict__ W,
                             int rowStride, int numKt, int numNt) {
  int gid = blockIdx.x * blockDim.x + threadIdx.x;
  int total = numKt * numNt * 512;
  if (gid >= total) return;
  int tile   = gid >> 9;         // (kt*numNt + nt)
  int within = gid & 511;
  int l = within >> 4;
  int j = within & 15;
  int kt = tile / numNt, nt = tile % numNt;
  int n = nt * 16 + (l & 15);
  int k = kt * 32 + (l >> 4) * 8 + (j & 7) + (j >> 3) * 16;
  dst[gid] = (_Float16)W[(long)n * rowStride + k];
}

__device__ __forceinline__ v16h loadA(const _Float16* rp) {
  // A layout: lane = M(0..15) in each 16-lane half; halves 0..7 = K+0..7,
  // halves 8..15 = K+16..23 (caller supplies K base = kt*32 + (lane>>4)*8).
  v8h lo = *(const v8h*)rp;
  v8h hi = *(const v8h*)(rp + 16);
  v16h a;
#pragma unroll
  for (int i = 0; i < 8; ++i) { a[i] = lo[i]; a[8 + i] = hi[i]; }
  return a;
}

__global__ __launch_bounds__(256)
void edge_mlp_scatter(const _Float16* __restrict__ xh,
                      const int*   __restrict__ eidx,
                      const float* __restrict__ eattr,
                      const float* __restrict__ W1,
                      const float* __restrict__ b1,
                      const float* __restrict__ b2,
                      const _Float16* __restrict__ pW1,
                      const _Float16* __restrict__ pW2,
                      float* __restrict__ aggr) {
  __shared__ _Float16 ldsIn[TILE_E * LSTRIDE];
  __shared__ _Float16 ldsH [TILE_E * LSTRIDE];
  __shared__ int   sdst[TILE_E];
  __shared__ float sea [TILE_E];
  __shared__ float sb1[HD], sw1c[HD], sb2[ND];

  const int t    = threadIdx.x;
  const int lane = t & 31;
  const int wid  = t >> 5;
  const long eBase = (long)blockIdx.x * TILE_E;

  if (t < TILE_E) {
    long e = eBase + t;
    if (e < NEDGES) { sdst[t] = eidx[(long)NEDGES + e]; sea[t] = eattr[e]; }
    else            { sdst[t] = -1;                     sea[t] = 0.f; }
  }
  if (t < HD) { sb1[t] = b1[t]; sw1c[t] = W1[(long)t * (2*ND+ED) + 2*ND]; }
  if (t < ND) { sb2[t] = b2[t]; }

  { // stage [xh[src] | xh[dst]] rows: async global->LDS, no VGPR round-trip
    int er = t >> 1;      // edge row within tile
    int hf = t & 1;       // 0: src half, 1: dst half
    long e = eBase + er;
    _Float16* drow = &ldsIn[er * LSTRIDE + hf * ND];
    if (e < NEDGES) {
      int idx = eidx[(long)hf * NEDGES + e];
      const _Float16* srow = xh + (long)idx * ND;
#pragma unroll
      for (int i = 0; i < ND / 8; ++i)
        async_ld16(drow + i * 8, srow + i * 8);
    } else {
      for (int i = 0; i < ND; ++i) drow[i] = (_Float16)0.f;
    }
  }
  wait_async();
  __syncthreads();

  const int rowBase = wid * 16;
  const int mLane = rowBase + (lane & 15);
  const int kHalf = (lane >> 4) * 8;
  const int col   = lane & 15;
  const int mAcc  = rowBase + (lane >> 4) * 8;   // + v -> output row

  float eav[8]; int dstv[8];
#pragma unroll
  for (int v = 0; v < 8; ++v) { eav[v] = sea[mAcc + v]; dstv[v] = sdst[mAcc + v]; }

  v16h a1[4];
#pragma unroll
  for (int kt = 0; kt < 4; ++kt)
    a1[kt] = loadA(&ldsIn[mLane * LSTRIDE + kt * 32 + kHalf]);

  // ---- layer 1: [128 rows] x K128 -> 128 cols; C init = b1 + ea * w1c ----
#pragma unroll
  for (int nt = 0; nt < 8; ++nt) {
    int c = nt * 16 + col;
    float bb = sb1[c], wc = sw1c[c];
    v8f acc;
#pragma unroll
    for (int v = 0; v < 8; ++v) acc[v] = bb + eav[v] * wc;
#pragma unroll
    for (int kt = 0; kt < 4; ++kt) {
      v16h b = *(const v16h*)&pW1[(((long)kt * 8 + nt) * 32 + lane) * 16];
      acc = wmma16(a1[kt], b, acc);
    }
#pragma unroll
    for (int v = 0; v < 8; ++v) {
      float h = acc[v] > 0.f ? acc[v] : 0.f;      // relu
      ldsH[(mAcc + v) * LSTRIDE + c] = (_Float16)h;
    }
  }

  // layer-2 A rows are this wave's own layer-1 rows: in-order LDS, no barrier
  v16h a2[4];
#pragma unroll
  for (int kt = 0; kt < 4; ++kt)
    a2[kt] = loadA(&ldsH[mLane * LSTRIDE + kt * 32 + kHalf]);

  // ---- layer 2: K128 -> 64 cols; then scatter-add to aggr[dst] ----
#pragma unroll
  for (int nt = 0; nt < 4; ++nt) {
    int c = nt * 16 + col;
    float bb = sb2[c];
    v8f acc;
#pragma unroll
    for (int v = 0; v < 8; ++v) acc[v] = bb;
#pragma unroll
    for (int kt = 0; kt < 4; ++kt) {
      v16h b = *(const v16h*)&pW2[(((long)kt * 4 + nt) * 32 + lane) * 16];
      acc = wmma16(a2[kt], b, acc);
    }
#pragma unroll
    for (int v = 0; v < 8; ++v) {
      int d = dstv[v];
      if (d >= 0) unsafeAtomicAdd(&aggr[(long)d * ND + c], acc[v]);
    }
  }
}

__global__ __launch_bounds__(256)
void gru_update(const float* __restrict__ aggr,
                const float* __restrict__ x,
                const _Float16* __restrict__ xh,
                const float* __restrict__ b_ih,
                const float* __restrict__ b_hh,
                const _Float16* __restrict__ pWih,
                const _Float16* __restrict__ pWhh,
                float* __restrict__ out) {
  __shared__ _Float16 ldsA[TILE_N * GSTRIDE];
  __shared__ _Float16 ldsX[TILE_N * GSTRIDE];

  const int t = threadIdx.x, lane = t & 31, wid = t >> 5;
  const long nBase = (long)blockIdx.x * TILE_N;

  {
    int r = t & 127, which = t >> 7;   // 128 threads for aggr, 128 for x
    long node = nBase + r;
    if (which) {                       // x rows: already f16, async copy
      _Float16* drow = ldsX + r * GSTRIDE;
      if (node < NNODES) {
        const _Float16* srow = xh + node * ND;
#pragma unroll
        for (int i = 0; i < ND / 8; ++i)
          async_ld16(drow + i * 8, srow + i * 8);
      } else {
        for (int i = 0; i < ND; ++i) drow[i] = (_Float16)0.f;
      }
    } else {                           // aggr rows: f32 -> f16 convert
      _Float16* drow = ldsA + r * GSTRIDE;
      if (node < NNODES) {
        const float4* sp = (const float4*)(aggr + node * ND);
#pragma unroll
        for (int i = 0; i < ND / 4; ++i) {
          float4 v = sp[i];
          drow[i*4+0] = (_Float16)v.x; drow[i*4+1] = (_Float16)v.y;
          drow[i*4+2] = (_Float16)v.z; drow[i*4+3] = (_Float16)v.w;
        }
      } else {
        for (int i = 0; i < ND; ++i) drow[i] = (_Float16)0.f;
      }
    }
  }
  wait_async();
  __syncthreads();

  const int rowBase = wid * 16;
  const int mLane = rowBase + (lane & 15);
  const int kHalf = (lane >> 4) * 8;
  const int col   = lane & 15;
  const int mAcc  = rowBase + (lane >> 4) * 8;

  v16h aA[2], aX[2];
#pragma unroll
  for (int kt = 0; kt < 2; ++kt) {
    aA[kt] = loadA(&ldsA[mLane * GSTRIDE + kt * 32 + kHalf]);
    aX[kt] = loadA(&ldsX[mLane * GSTRIDE + kt * 32 + kHalf]);
  }

  // N=192 = 12 n-tiles: r -> jt, z -> jt+4, n -> jt+8 (aligned lane slots)
#pragma unroll
  for (int jt = 0; jt < 4; ++jt) {
    int c = jt * 16 + col;
    float bir = b_ih[c], biz = b_ih[ND + c], bin = b_ih[2*ND + c];
    float bhr = b_hh[c], bhz = b_hh[ND + c], bhn = b_hh[2*ND + c];
    v8f ir, iz, inn, hr, hz, hn;
#pragma unroll
    for (int v = 0; v < 8; ++v) {
      ir[v] = bir; iz[v] = biz; inn[v] = bin;
      hr[v] = bhr; hz[v] = bhz; hn[v]  = bhn;
    }
#pragma unroll
    for (int kt = 0; kt < 2; ++kt) {
      v16h br = *(const v16h*)&pWih[(((long)kt*12 + jt    ) * 32 + lane) * 16];
      v16h bz = *(const v16h*)&pWih[(((long)kt*12 + jt + 4) * 32 + lane) * 16];
      v16h bn = *(const v16h*)&pWih[(((long)kt*12 + jt + 8) * 32 + lane) * 16];
      ir  = wmma16(aA[kt], br, ir);
      iz  = wmma16(aA[kt], bz, iz);
      inn = wmma16(aA[kt], bn, inn);
      br = *(const v16h*)&pWhh[(((long)kt*12 + jt    ) * 32 + lane) * 16];
      bz = *(const v16h*)&pWhh[(((long)kt*12 + jt + 4) * 32 + lane) * 16];
      bn = *(const v16h*)&pWhh[(((long)kt*12 + jt + 8) * 32 + lane) * 16];
      hr = wmma16(aX[kt], br, hr);
      hz = wmma16(aX[kt], bz, hz);
      hn = wmma16(aX[kt], bn, hn);
    }
#pragma unroll
    for (int v = 0; v < 8; ++v) {
      long node = nBase + mAcc + v;
      if (node < NNODES) {
        float r = fast_sigmoid(ir[v] + hr[v]);
        float z = fast_sigmoid(iz[v] + hz[v]);
        float n = fast_tanh(inn[v] + r * hn[v]);
        float xv = x[node * ND + c];
        out[node * ND + c] = (1.f - z) * n + z * xv;
      }
    }
  }
}

extern "C" void kernel_launch(void* const* d_in, const int* in_sizes, int n_in,
                              void* d_out, int out_size, void* d_ws, size_t ws_size,
                              hipStream_t stream) {
  const float* x     = (const float*)d_in[0];
  const int*   eidx  = (const int*)  d_in[1];
  const float* eattr = (const float*)d_in[2];
  const float* W1    = (const float*)d_in[3];
  const float* b1    = (const float*)d_in[4];
  const float* W2    = (const float*)d_in[5];
  const float* b2    = (const float*)d_in[6];
  const float* Wih   = (const float*)d_in[7];
  const float* bih   = (const float*)d_in[8];
  const float* Whh   = (const float*)d_in[9];
  const float* bhh   = (const float*)d_in[10];
  float* out = (float*)d_out;

  char* ws = (char*)d_ws;
  auto align256 = [](size_t o) { return (o + 255) & ~(size_t)255; };
  float* aggr = (float*)ws;
  size_t off = align256((size_t)NNODES * ND * sizeof(float));
  _Float16* xh   = (_Float16*)(ws + off);
  off = align256(off + (size_t)NNODES * ND * sizeof(_Float16));
  _Float16* pW1  = (_Float16*)(ws + off); off = align256(off + 4*8*512*2);
  _Float16* pW2  = (_Float16*)(ws + off); off = align256(off + 4*4*512*2);
  _Float16* pWih = (_Float16*)(ws + off); off = align256(off + 2*12*512*2);
  _Float16* pWhh = (_Float16*)(ws + off);

  hipMemsetAsync(aggr, 0, (size_t)NNODES * ND * sizeof(float), stream);
  cvt_x_f16<<<(NNODES * ND / 4 + 255) / 256, 256, 0, stream>>>(xh, x);
  pack_weights<<<(4*8*512 + 255)/256, 256, 0, stream>>>(pW1,  W1,  2*ND+ED, 4, 8);
  pack_weights<<<(4*4*512 + 255)/256, 256, 0, stream>>>(pW2,  W2,  HD,      4, 4);
  pack_weights<<<(2*12*512 + 255)/256, 256, 0, stream>>>(pWih, Wih, ND,     2, 12);
  pack_weights<<<(2*12*512 + 255)/256, 256, 0, stream>>>(pWhh, Whh, ND,     2, 12);

  edge_mlp_scatter<<<(NEDGES + TILE_E - 1) / TILE_E, 256, 0, stream>>>(
      xh, eidx, eattr, W1, b1, b2, pW1, pW2, aggr);
  gru_update<<<(NNODES + TILE_N - 1) / TILE_N, 256, 0, stream>>>(
      aggr, x, xh, bih, bhh, pWih, pWhh, out);
}
